// SparseMoE_16286515987179
// MI455X (gfx1250) — matrix operand order repaired and checked by reading, not compiled, not run on hardware
//
#include <hip/hip_runtime.h>
#include <cstdint>

#define T_TOK 4096
#define HID   1024
#define INTER 4096
#define NEXP  8
#define CAP   4128     // per-expert list capacity (>= 4096, padded)
#define MT    32       // tokens per MoE block
#define IC    64       // inter-dim chunk

typedef __attribute__((ext_vector_type(16))) __bf16   v16bf;
typedef __attribute__((ext_vector_type(8)))  float    v8f;
typedef __attribute__((ext_vector_type(4)))  unsigned int v4u;

union Frag { v16bf v; v4u u[2]; };

__device__ __forceinline__ unsigned short f2bf(float f) {
  unsigned u = __builtin_bit_cast(unsigned, f);
  u += 0x7FFFu + ((u >> 16) & 1u);           // round-to-nearest-even
  return (unsigned short)(u >> 16);
}

// ---------------- workspace layout (bytes) ----------------
// w1t bf16 [E][I][H] : 0            .. 67108864
// w2t bf16 [E][H][I] : 67108864     .. 134217728
// xb  bf16 [T+32][H] : 134217728    .. 142671872
// counts int[8]      : 142671872    (256 B slot)
// toklist int[E][CAP]: 142672128    (132096 B)
// wtlist  f32[E][CAP]: 142804224    (132096 B)
// y2  f32 [2T+32][H] : 142936320    (33685504 B)   total ~176.6 MB
#define O_W1T 0ull
#define O_W2T 67108864ull
#define O_XB  134217728ull
#define O_CNT 142671872ull
#define O_TOK 142672128ull
#define O_WT  142804224ull
#define O_Y2  142936320ull

// ---------------- small helpers ----------------
__global__ void zero_counts(int* counts) {
  if (threadIdx.x < NEXP) counts[threadIdx.x] = 0;
}

__global__ void cvt_x(const float* __restrict__ in, unsigned short* __restrict__ out, int n8) {
  int i = blockIdx.x * 256 + threadIdx.x;
  if (i >= n8) return;
  const float4* p = (const float4*)in + (size_t)i * 2;
  float4 f0 = p[0], f1 = p[1];
  union { v4u u; unsigned short s[8]; } r;
  r.s[0] = f2bf(f0.x); r.s[1] = f2bf(f0.y); r.s[2] = f2bf(f0.z); r.s[3] = f2bf(f0.w);
  r.s[4] = f2bf(f1.x); r.s[5] = f2bf(f1.y); r.s[6] = f2bf(f1.z); r.s[7] = f2bf(f1.w);
  ((v4u*)out)[i] = r.u;
}

// in [nmat][R][C] fp32 -> out [nmat][C][R] bf16, LDS-tiled for coalescing
__global__ void transpose_cvt(const float* __restrict__ in, unsigned short* __restrict__ out,
                              int R, int C) {
  __shared__ unsigned short tile[32][33];
  const float* src = in + (size_t)blockIdx.z * R * C;
  unsigned short* dst = out + (size_t)blockIdx.z * R * C;
  int c0 = blockIdx.x * 32, r0 = blockIdx.y * 32;
  #pragma unroll
  for (int i = 0; i < 4; ++i) {
    int r = r0 + threadIdx.y + i * 8;
    tile[threadIdx.y + i * 8][threadIdx.x] = f2bf(src[(size_t)r * C + c0 + threadIdx.x]);
  }
  __syncthreads();
  #pragma unroll
  for (int i = 0; i < 4; ++i) {
    int c = c0 + threadIdx.y + i * 8;
    dst[(size_t)c * R + r0 + threadIdx.x] = tile[threadIdx.x][threadIdx.y + i * 8];
  }
}

// ---------------- router: one wave32 per token ----------------
__global__ void router_kernel(const float* __restrict__ x, const float* __restrict__ gate,
                              float* __restrict__ logits_out, int* __restrict__ counts,
                              int* __restrict__ toklist, float* __restrict__ wtlist) {
  int wv = threadIdx.x >> 5, lane = threadIdx.x & 31;
  int t = blockIdx.x * 8 + wv;
  const float* xr = x + (size_t)t * HID;
  float acc[NEXP];
  #pragma unroll
  for (int j = 0; j < NEXP; ++j) acc[j] = 0.f;
  for (int hb = 0; hb < HID; hb += 32) {
    float xv = xr[hb + lane];
    const float4* g = (const float4*)(gate + (size_t)(hb + lane) * NEXP);
    float4 g0 = g[0], g1 = g[1];
    acc[0] += xv * g0.x; acc[1] += xv * g0.y; acc[2] += xv * g0.z; acc[3] += xv * g0.w;
    acc[4] += xv * g1.x; acc[5] += xv * g1.y; acc[6] += xv * g1.z; acc[7] += xv * g1.w;
  }
  #pragma unroll
  for (int j = 0; j < NEXP; ++j)
    #pragma unroll
    for (int s = 16; s > 0; s >>= 1) acc[j] += __shfl_xor(acc[j], s, 32);

  // softmax probs (only top-2 ratio matters for the weights)
  float mx = acc[0];
  #pragma unroll
  for (int j = 1; j < NEXP; ++j) mx = fmaxf(mx, acc[j]);
  float p[NEXP];
  #pragma unroll
  for (int j = 0; j < NEXP; ++j) p[j] = __expf(acc[j] - mx);
  int e0 = 0; float p0 = p[0];
  #pragma unroll
  for (int j = 1; j < NEXP; ++j) if (p[j] > p0) { p0 = p[j]; e0 = j; }
  int e1 = (e0 == 0) ? 1 : 0; float p1 = p[e1];
  #pragma unroll
  for (int j = 0; j < NEXP; ++j) if (j != e0 && p[j] > p1) { p1 = p[j]; e1 = j; }
  float inv = 1.f / (p0 + p1);

  if (lane == 0) {
    #pragma unroll
    for (int j = 0; j < NEXP; ++j) logits_out[(size_t)t * NEXP + j] = acc[j];
    int pos0 = atomicAdd(&counts[e0], 1);
    toklist[e0 * CAP + pos0] = (t << 1);
    wtlist[e0 * CAP + pos0] = p0 * inv;
    int pos1 = atomicAdd(&counts[e1], 1);
    toklist[e1 * CAP + pos1] = (t << 1) | 1;
    wtlist[e1 * CAP + pos1] = p1 * inv;
  }
}

__global__ void pad_lists(const int* __restrict__ counts, int* __restrict__ toklist,
                          float* __restrict__ wtlist) {
  int e = blockIdx.x;
  int c = counts[e], pd = (c + MT - 1) & ~(MT - 1);
  for (int i = c + threadIdx.x; i < pd; i += 32) {
    toklist[e * CAP + i] = (T_TOK << 1);   // sentinel -> dump row of y2
    wtlist[e * CAP + i] = 0.f;
  }
}

// ---------------- fused MoE GEMM1 -> GELU -> GEMM2 ----------------
__global__ void __launch_bounds__(256, 1)
moe_kernel(const unsigned short* __restrict__ xb, const unsigned short* __restrict__ w1t,
           const unsigned short* __restrict__ w2t, const int* __restrict__ counts,
           const int* __restrict__ toklist, const float* __restrict__ wtlist,
           float* __restrict__ y2) {
  int e = blockIdx.y;
  int pd = (counts[e] + MT - 1) & ~(MT - 1);
  int m0 = blockIdx.x * MT;
  if (m0 >= pd) return;

  __shared__ unsigned short sh1[MT * 80];   // h1 chunk, stride 80 bf16 (160B, 16B-aligned)
  __shared__ int   sent[MT];
  __shared__ float swt[MT];

  int tid = threadIdx.x;
  if (tid < MT) {
    sent[tid] = toklist[e * CAP + m0 + tid];
    swt[tid]  = wtlist[e * CAP + m0 + tid];
  }
  __syncthreads();

  int wv = tid >> 5, lane = tid & 31;
  int mh = wv >> 2;          // 16-row half of the 32-token tile
  int nq = wv & 3;           // GEMM1 16-col slot / GEMM2 256-col quarter
  int lo = lane & 15, hi = lane >> 4;

  const unsigned short* w1e = w1t + (size_t)e * INTER * HID;
  const unsigned short* w2e = w2t + (size_t)e * HID * INTER;

  // A-fragment source row for both GEMMs: token row mh*16 + lo
  int arow = mh * 16 + lo;
  const unsigned short* xr = xb + (size_t)(sent[arow] >> 1) * HID;

  v8f acc2[16];
  #pragma unroll
  for (int i = 0; i < 16; ++i)
    #pragma unroll
    for (int j = 0; j < 8; ++j) acc2[i][j] = 0.f;

  for (int ch = 0; ch < INTER; ch += IC) {
    // ---- GEMM1: h1[32,64] = x[32,1024] @ w1[:,ch:ch+64]; one 16x16 tile/wave ----
    v8f acc1;
    #pragma unroll
    for (int j = 0; j < 8; ++j) acc1[j] = 0.f;
    int n1 = ch + nq * 16 + lo;                    // inter index = B column
    #pragma unroll 4
    for (int k0 = 0; k0 < HID; k0 += 32) {
      Frag a, b;
      int ak = k0 + hi * 8;
      a.u[0] = *(const v4u*)&xr[ak];
      a.u[1] = *(const v4u*)&xr[ak + 16];
      int bk = k0 + hi * 8;
      b.u[0] = *(const v4u*)&w1e[(size_t)n1 * HID + bk];
      b.u[1] = *(const v4u*)&w1e[(size_t)n1 * HID + bk + 16];
      acc1 = __builtin_amdgcn_wmma_f32_16x16x32_bf16(false, a.v, false, b.v,
                                                     (short)0, acc1, false, false);
    }
    // exact GELU, pack bf16 -> LDS
    #pragma unroll
    for (int r = 0; r < 8; ++r) {
      float v = acc1[r];
      v = 0.5f * v * (1.f + erff(v * 0.70710678118654752f));
      sh1[(mh * 16 + r + hi * 8) * 80 + nq * 16 + lo] = f2bf(v);
    }
    __syncthreads();

    // ---- GEMM2: out[32,1024] += h1[32,64] @ w2[ch:ch+64,:]; wave owns 16x256 ----
    Frag a0, a1;
    int hb = (mh * 16 + lo) * 80 + hi * 8;
    a0.u[0] = *(const v4u*)&sh1[hb];
    a0.u[1] = *(const v4u*)&sh1[hb + 16];
    a1.u[0] = *(const v4u*)&sh1[hb + 32];
    a1.u[1] = *(const v4u*)&sh1[hb + 32 + 16];
    #pragma unroll
    for (int nt = 0; nt < 16; ++nt) {
      int n = nq * 256 + nt * 16 + lo;             // hidden index = B column
      const unsigned short* wr = &w2e[(size_t)n * INTER + ch + hi * 8];
      Frag b0, b1;
      b0.u[0] = *(const v4u*)&wr[0];
      b0.u[1] = *(const v4u*)&wr[16];
      b1.u[0] = *(const v4u*)&wr[32];
      b1.u[1] = *(const v4u*)&wr[48];
      acc2[nt] = __builtin_amdgcn_wmma_f32_16x16x32_bf16(false, a0.v, false, b0.v,
                                                         (short)0, acc2[nt], false, false);
      acc2[nt] = __builtin_amdgcn_wmma_f32_16x16x32_bf16(false, a1.v, false, b1.v,
                                                         (short)0, acc2[nt], false, false);
    }
    __syncthreads();
  }

  // scale by routing weight, scatter to per-(token,slot) slab (no atomics)
  #pragma unroll
  for (int nt = 0; nt < 16; ++nt) {
    int col = nq * 256 + nt * 16 + lo;
    #pragma unroll
    for (int r = 0; r < 8; ++r) {
      int row = mh * 16 + r + hi * 8;
      y2[(size_t)sent[row] * HID + col] = acc2[nt][r] * swt[row];
    }
  }
}

__global__ void combine_kernel(const float* __restrict__ y2, float* __restrict__ out) {
  int i = blockIdx.x * 256 + threadIdx.x;          // over T*H/4
  int t = i >> 8, q = i & 255;
  float4 a = ((const float4*)(y2 + (size_t)(2 * t) * HID))[q];
  float4 b = ((const float4*)(y2 + (size_t)(2 * t + 1) * HID))[q];
  float4 r; r.x = a.x + b.x; r.y = a.y + b.y; r.z = a.z + b.z; r.w = a.w + b.w;
  ((float4*)out)[i] = r;
}

// ---------------- launch ----------------
extern "C" void kernel_launch(void* const* d_in, const int* in_sizes, int n_in,
                              void* d_out, int out_size, void* d_ws, size_t ws_size,
                              hipStream_t stream) {
  const float* x    = (const float*)d_in[0];
  const float* gate = (const float*)d_in[1];
  const float* w1   = (const float*)d_in[2];
  const float* w2   = (const float*)d_in[3];
  float* out    = (float*)d_out;
  float* logits = out + (size_t)T_TOK * HID;

  char* ws = (char*)d_ws;
  unsigned short* w1t = (unsigned short*)(ws + O_W1T);
  unsigned short* w2t = (unsigned short*)(ws + O_W2T);
  unsigned short* xb  = (unsigned short*)(ws + O_XB);
  int*   counts = (int*)(ws + O_CNT);
  int*   tok    = (int*)(ws + O_TOK);
  float* wt     = (float*)(ws + O_WT);
  float* y2     = (float*)(ws + O_Y2);

  zero_counts<<<1, 32, 0, stream>>>(counts);
  cvt_x<<<(T_TOK * HID / 8 + 255) / 256, 256, 0, stream>>>(x, xb, T_TOK * HID / 8);
  transpose_cvt<<<dim3(INTER / 32, HID / 32, NEXP), dim3(32, 8), 0, stream>>>(w1, w1t, HID, INTER);
  transpose_cvt<<<dim3(HID / 32, INTER / 32, NEXP), dim3(32, 8), 0, stream>>>(w2, w2t, INTER, HID);
  router_kernel<<<T_TOK / 8, 256, 0, stream>>>(x, gate, logits, counts, tok, wt);
  pad_lists<<<NEXP, 32, 0, stream>>>(counts, tok, wt);
  moe_kernel<<<dim3(T_TOK / MT, NEXP), 256, 0, stream>>>(xb, w1t, w2t, counts, tok, wt, y2);
  combine_kernel<<<T_TOK * HID / 4 / 256, 256, 0, stream>>>(y2, out);
}
